// GNN_79937931313503
// MI455X (gfx1250) — matrix operand order repaired and checked by reading, not compiled, not run on hardware
//
#include <hip/hip_runtime.h>
#include <math.h>

// ---------------- CDNA5 WMMA types ----------------
typedef __attribute__((ext_vector_type(16))) __bf16 v16bf;
typedef __attribute__((ext_vector_type(8)))  float  v8f;

#define NN  8192
#define DIN 128
#define HH  256

// Async global->LDS (CDNA5 ASYNCcnt path), guarded so compile can't hard-fail.
#if defined(__has_builtin)
#if __has_builtin(__builtin_amdgcn_global_load_async_to_lds_b128) && \
    __has_builtin(__builtin_amdgcn_s_wait_asynccnt)
#define HAVE_ASYNC_LDS 1
#endif
#endif
#ifndef HAVE_ASYNC_LDS
#define HAVE_ASYNC_LDS 0
#endif

#if HAVE_ASYNC_LDS
// Builtin params are pointers to a 16-byte int vector (per hipcc diagnostic),
// global (AS1) source and LDS (AS3) destination.
typedef int vi4 __attribute__((vector_size(4 * sizeof(int))));
typedef __attribute__((address_space(1))) vi4 as1_vi4;
typedef __attribute__((address_space(3))) vi4 as3_vi4;
#endif

__device__ __forceinline__ unsigned short f2b(float f) {
  unsigned int u = __float_as_uint(f);
  u += 0x7FFFu + ((u >> 16) & 1u);          // round-to-nearest-even
  return (unsigned short)(u >> 16);
}
__device__ __forceinline__ float b2f(unsigned short b) {
  return __uint_as_float(((unsigned int)b) << 16);
}

union Frag {
  v16bf v;
  unsigned short us[16];
  uint4 q[2];
};

enum { F_ABS = 1, F_BIAS = 4, F_ACC = 8, F_RES = 16, F_RELU = 32, F_OUTF32 = 64 };

// ---------------- generic WMMA bf16 GEMM (dense operands) ----------------
// C[M x Nc] = A[M x K] (row-major bf16) @ BT[Nc x K]^T (BT row-major bf16)
template <int FLAGS>
__global__ __launch_bounds__(256) void wmma_gemm_k(
    const unsigned short* __restrict__ A, const unsigned short* __restrict__ BT,
    int M, int Nc, int K,
    const float* __restrict__ bias, const float* __restrict__ accIn,
    const unsigned short* __restrict__ res,
    unsigned short* __restrict__ outB, float* __restrict__ outF) {
  const int wave = threadIdx.x >> 5;
  const int lane = threadIdx.x & 31;
  const int half = lane >> 4;
  const int l16  = lane & 15;
  const int col0 = blockIdx.x * 128 + wave * 16;
  const int row0 = blockIdx.y * 16;

  // A fragment as loaded here (same k order as B): lane L holds row M=L%16, K = (L/16)*16 + 0..15 (contiguous)
  const unsigned short* Ap = A + (size_t)(row0 + l16) * K + half * 16;
  // B fragment as loaded here (same k order as A): lane L holds col N=L%16, halves K = (L/16)*16 + 0..15 (contiguous)
  const unsigned short* Bp = BT + (size_t)(col0 + l16) * K + half * 16;

  v8f c = {};
  for (int kb = 0; kb < K; kb += 32) {
    Frag a, b;
    a.q[0] = *(const uint4*)(Ap + kb);
    a.q[1] = *(const uint4*)(Ap + kb + 8);
    b.q[0] = *(const uint4*)(Bp + kb);
    b.q[1] = *(const uint4*)(Bp + kb + 8);
    c = __builtin_amdgcn_wmma_f32_16x16x32_bf16(false, a.v, false, b.v,
                                                (short)0, c, false, false);
  }

  const int col = col0 + l16;
#pragma unroll
  for (int r = 0; r < 8; ++r) {
    const int row = row0 + r + half * 8;
    float v = c[r];
    if (FLAGS & F_ABS)  v = fabsf(v);
    if (FLAGS & F_BIAS) v += bias[col];
    if (FLAGS & F_ACC)  v += accIn[(size_t)row * Nc + col];
    if (FLAGS & F_RES)  v += b2f(res[(size_t)row * Nc + col]);
    if (FLAGS & F_RELU) v = fmaxf(v, 0.0f);
    if (FLAGS & F_OUTF32) outF[(size_t)row * Nc + col] = v;
    else                  outB[(size_t)row * Nc + col] = f2b(v);
  }
}

// ---------------- adjacency GEMM: dense Abin (0/1 bf16) with LDS-staged A ----------------
// out[M x Nc] = (Abin @ BT^T) * degInv[row]; Abin bf16 [M x K], L2-resident.
// Block tile: 64 rows x 128 cols, 8 waves; each wave: 4 row tiles x 16 cols.
// A chunks (64 x 64 bf16 = 8 KB) double-buffered in LDS, staged via the CDNA5
// async global->LDS engine (ASYNCcnt) when available. Hot loop has no per-element VALU.
#define KC    64
#define AROWS 64
#define APAD  8
__global__ __launch_bounds__(256) void adj_gemm_lds_k(
    const unsigned short* __restrict__ Abin, const unsigned short* __restrict__ BT,
    int Nc, int K,
    const float* __restrict__ degInv,
    unsigned short* __restrict__ outB) {
  __shared__ unsigned short abuf[2][AROWS][KC + APAD];

  const int tid  = threadIdx.x;
  const int wave = tid >> 5;
  const int lane = tid & 31;
  const int half = lane >> 4;
  const int l16  = lane & 15;
  const int col0 = blockIdx.x * 128 + wave * 16;
  const int row0 = blockIdx.y * AROWS;

  // staging: 512 x 16B segments per chunk; thread t does rows (t>>3) and (t>>3)+32
  const int srow = tid >> 3;
  const int soff = (tid & 7) * 8;

  auto stage = [&](int kb, int bufi) {
    const unsigned short* g0 = Abin + (size_t)(row0 + srow) * K + kb + soff;
    const unsigned short* g1 = Abin + (size_t)(row0 + srow + 32) * K + kb + soff;
#if HAVE_ASYNC_LDS
    __builtin_amdgcn_global_load_async_to_lds_b128(
        (as1_vi4*)g0, (as3_vi4*)&abuf[bufi][srow][soff], 0, 0);
    __builtin_amdgcn_global_load_async_to_lds_b128(
        (as1_vi4*)g1, (as3_vi4*)&abuf[bufi][srow + 32][soff], 0, 0);
#else
    const uint4 d0 = *(const uint4*)g0;
    const uint4 d1 = *(const uint4*)g1;
    *(uint4*)&abuf[bufi][srow][soff]      = d0;
    *(uint4*)&abuf[bufi][srow + 32][soff] = d1;
#endif
  };

  const unsigned short* Bp = BT + (size_t)(col0 + l16) * K + half * 16;
  v8f c[4];
#pragma unroll
  for (int rt = 0; rt < 4; ++rt) c[rt] = (v8f){};

  const int nch = K / KC;
  stage(0, 0);

  for (int ch = 0; ch < nch; ++ch) {
    const int cur = ch & 1;
    if (ch + 1 < nch) {
      stage((ch + 1) * KC, cur ^ 1);
#if HAVE_ASYNC_LDS
      __builtin_amdgcn_s_wait_asynccnt(2);   // chunk ch landed; next chunk in flight
#endif
    } else {
#if HAVE_ASYNC_LDS
      __builtin_amdgcn_s_wait_asynccnt(0);
#endif
    }
    __syncthreads();                         // chunk ch visible to all waves

    const int kb = ch * KC;
#pragma unroll
    for (int kk = 0; kk < KC; kk += 32) {
      Frag b;
      b.q[0] = *(const uint4*)(Bp + kb + kk);
      b.q[1] = *(const uint4*)(Bp + kb + kk + 8);
      const int k0 = kk + half * 16;
#pragma unroll
      for (int rt = 0; rt < 4; ++rt) {       // reuse B fragment across 4 row tiles
        Frag a;
        a.q[0] = *(const uint4*)&abuf[cur][rt * 16 + l16][k0];
        a.q[1] = *(const uint4*)&abuf[cur][rt * 16 + l16][k0 + 8];
        c[rt] = __builtin_amdgcn_wmma_f32_16x16x32_bf16(false, a.v, false, b.v,
                                                        (short)0, c[rt], false, false);
      }
    }
    __syncthreads();                         // all waves done before buffer reuse
  }

  const int col = col0 + l16;
#pragma unroll
  for (int rt = 0; rt < 4; ++rt) {
#pragma unroll
    for (int r = 0; r < 8; ++r) {
      const int row = row0 + rt * 16 + r + half * 8;
      outB[(size_t)row * Nc + col] = f2b(c[rt][r] * degInv[row]);
    }
  }
}

// ---------------- helpers ----------------
__global__ void rownorm_k(const float* __restrict__ x, unsigned short* __restrict__ xb,
                          unsigned short* __restrict__ xnb) {
  const int row = blockIdx.x, t = threadIdx.x;  // 128 threads
  const float v = x[(size_t)row * DIN + t];
  __shared__ float s[DIN];
  s[t] = v * v;
  __syncthreads();
  for (int o = DIN / 2; o > 0; o >>= 1) {
    if (t < o) s[t] += s[t + o];
    __syncthreads();
  }
  const float inv = 1.0f / fmaxf(sqrtf(s[0]), 1e-8f);
  xb [(size_t)row * DIN + t] = f2b(v);
  xnb[(size_t)row * DIN + t] = f2b(v * inv);
}

__global__ void convt_w_k(const float* __restrict__ in, unsigned short* __restrict__ out,
                          int K, int N) {
  const int idx = blockIdx.x * blockDim.x + threadIdx.x;
  if (idx >= K * N) return;
  const int n = idx % N, k = idx / N;
  out[(size_t)n * K + k] = f2b(in[(size_t)k * N + n]);
}

__global__ void transpose_k(const unsigned short* __restrict__ in,
                            unsigned short* __restrict__ out, int M, int N) {
  __shared__ unsigned short t[32][33];
  const int nb = blockIdx.x * 32, mb = blockIdx.y * 32;
  for (int i = threadIdx.y; i < 32; i += 8)
    t[i][threadIdx.x] = in[(size_t)(mb + i) * N + (nb + threadIdx.x)];
  __syncthreads();
  for (int i = threadIdx.y; i < 32; i += 8)
    out[(size_t)(nb + i) * M + (mb + threadIdx.x)] = t[threadIdx.x][i];
}

__global__ void zero_u32_k(unsigned int* p, int n) {
  const int i = blockIdx.x * blockDim.x + threadIdx.x;
  if (i < n) p[i] = 0;
}

// Abin = (sim >= eps) ? bf16(1.0) : 0   -- one memory-bound pass, uint4-vectorized
__global__ void binarize_k(const unsigned short* __restrict__ sim,
                           unsigned short* __restrict__ abin,
                           const unsigned int* __restrict__ epsB, unsigned long long n) {
  const unsigned short e = (unsigned short)epsB[0];
  const size_t i0 = ((size_t)blockIdx.x * blockDim.x + threadIdx.x) * 8;
  const size_t stride = (size_t)gridDim.x * blockDim.x * 8;
  for (size_t i = i0; i < n; i += stride) {
    union { uint4 q; unsigned short us[8]; } u;
    u.q = *(const uint4*)(sim + i);
#pragma unroll
    for (int j = 0; j < 8; ++j)
      u.us[j] = (u.us[j] >= e) ? (unsigned short)0x3F80u : (unsigned short)0u;
    *(uint4*)(abin + i) = u.q;
  }
}

__global__ void hist_k(const unsigned short* __restrict__ sim, unsigned long long n,
                       unsigned int* __restrict__ gh) {
  __shared__ unsigned int lh[16384];
  for (int i = threadIdx.x; i < 16384; i += 256) lh[i] = 0;
  __syncthreads();
  const size_t stride = (size_t)gridDim.x * 256;
  for (size_t i = (size_t)blockIdx.x * 256 + threadIdx.x; i < n; i += stride)
    atomicAdd(&lh[sim[i] >> 2], 1u);
  __syncthreads();
  for (int i = threadIdx.x; i < 16384; i += 256)
    if (lh[i]) atomicAdd(&gh[i], lh[i]);
}

__global__ void coarse_scan_k(const unsigned int* __restrict__ gh, unsigned int* sel,
                              unsigned long long krank) {
  unsigned long long cum = 0;
  for (int b = 0; b < 16384; ++b) {
    const unsigned long long c = gh[b];
    if (cum + c > krank) { sel[0] = (unsigned int)b; sel[1] = (unsigned int)(krank - cum); return; }
    cum += c;
  }
  sel[0] = 16383u; sel[1] = 0u;
}

__global__ void refine_count_k(const unsigned short* __restrict__ sim, unsigned long long n,
                               const unsigned int* __restrict__ sel, unsigned int* __restrict__ rc) {
  __shared__ unsigned int lc[4];
  if (threadIdx.x < 4) lc[threadIdx.x] = 0;
  __syncthreads();
  const unsigned int bin = sel[0];
  const size_t stride = (size_t)gridDim.x * 256;
  for (size_t i = (size_t)blockIdx.x * 256 + threadIdx.x; i < n; i += stride) {
    const unsigned short v = sim[i];
    if ((unsigned int)(v >> 2) == bin) atomicAdd(&lc[v & 3], 1u);
  }
  __syncthreads();
  if (threadIdx.x < 4 && lc[threadIdx.x]) atomicAdd(&rc[threadIdx.x], lc[threadIdx.x]);
}

__global__ void final_eps_k(const unsigned int* __restrict__ sel,
                            const unsigned int* __restrict__ rc, unsigned int* epsOut) {
  const unsigned int bin = sel[0], r = sel[1];
  unsigned int cum = 0, bits = bin << 2;
  for (int j = 0; j < 4; ++j) {
    cum += rc[j];
    if (cum > r) { bits = (bin << 2) | (unsigned int)j; break; }
  }
  epsOut[0] = bits;
}

__global__ void deginv_k(const unsigned short* __restrict__ sim,
                         const unsigned int* __restrict__ epsB,
                         float* __restrict__ degInv, int N) {
  const int row = blockIdx.x, t = threadIdx.x;
  const unsigned short e = (unsigned short)epsB[0];
  int cnt = 0;
  for (int j = t; j < N; j += 256) cnt += (sim[(size_t)row * N + j] >= e) ? 1 : 0;
  __shared__ int s[256];
  s[t] = cnt;
  __syncthreads();
  for (int o = 128; o > 0; o >>= 1) {
    if (t < o) s[t] += s[t + o];
    __syncthreads();
  }
  if (t == 0) degInv[row] = 1.0f / fmaxf((float)s[0], 1.0f);
}

__global__ void out1_k(const unsigned short* __restrict__ magg, const unsigned short* __restrict__ act,
                       const unsigned short* __restrict__ wl, const unsigned short* __restrict__ wr,
                       const unsigned short* __restrict__ wsc,
                       const float* __restrict__ o1_bl, const float* __restrict__ osc_b,
                       float* __restrict__ oval, float* __restrict__ scv) {
  const int row = blockIdx.x, t = threadIdx.x;  // 256 threads
  const float m = b2f(magg[(size_t)row * HH + t]);
  const float h = b2f(act [(size_t)row * HH + t]);
  __shared__ float sa[HH], sb[HH], sc_[HH];
  sa[t] = m * b2f(wl[t]);
  sb[t] = h * b2f(wr[t]);
  sc_[t] = h * b2f(wsc[t]);
  __syncthreads();
  for (int o = 128; o > 0; o >>= 1) {
    if (t < o) { sa[t] += sa[t + o]; sb[t] += sb[t + o]; sc_[t] += sc_[t + o]; }
    __syncthreads();
  }
  if (t == 0) {
    oval[row] = fmaxf(sa[0] + o1_bl[0] + sb[0], 0.0f);
    scv[row]  = sc_[0] + osc_b[0];
  }
}

__global__ void final_k(const unsigned short* __restrict__ abin, const float* __restrict__ oval,
                        const float* __restrict__ degInv,
                        const float* __restrict__ o2_wl, const float* __restrict__ o2_bl,
                        const float* __restrict__ o2_wr, const float* __restrict__ scv,
                        float* __restrict__ out, int N) {
  const int row = blockIdx.x, t = threadIdx.x;
  float s = 0.0f;
  for (int j = t; j < N; j += 256)
    if (abin[(size_t)row * N + j]) s += oval[j];
  __shared__ float sh[256];
  sh[t] = s;
  __syncthreads();
  for (int o = 128; o > 0; o >>= 1) {
    if (t < o) sh[t] += sh[t + o];
    __syncthreads();
  }
  if (t == 0) {
    const float agg = sh[0] * degInv[row];
    const float z = agg * o2_wl[0] + o2_bl[0] + oval[row] * o2_wr[0] + scv[row];
    out[row] = 1.0f / (1.0f + expf(-z));
  }
}

// ---------------- host launcher ----------------
extern "C" void kernel_launch(void* const* d_in, const int* in_sizes, int n_in,
                              void* d_out, int out_size, void* d_ws, size_t ws_size,
                              hipStream_t stream) {
  (void)in_sizes; (void)n_in; (void)out_size; (void)ws_size;
  const float* x     = (const float*)d_in[0];
  const float* w_in  = (const float*)d_in[1];
  const float* b_in  = (const float*)d_in[2];
  const float* h1_wl = (const float*)d_in[3];
  const float* h1_bl = (const float*)d_in[4];
  const float* h1_wr = (const float*)d_in[5];
  const float* h2_wl = (const float*)d_in[6];
  const float* h2_bl = (const float*)d_in[7];
  const float* h2_wr = (const float*)d_in[8];
  const float* o1_wl = (const float*)d_in[9];
  const float* o1_bl = (const float*)d_in[10];
  const float* o1_wr = (const float*)d_in[11];
  const float* o2_wl = (const float*)d_in[12];
  const float* o2_bl = (const float*)d_in[13];
  const float* o2_wr = (const float*)d_in[14];
  const float* osc_w = (const float*)d_in[15];
  const float* osc_b = (const float*)d_in[16];
  float* out = (float*)d_out;

  char* base = (char*)d_ws;
  size_t cur = 0;
  auto alloc = [&](size_t bytes) -> void* {
    void* p = base + cur;
    cur += (bytes + 255) & ~(size_t)255;
    return p;
  };
  unsigned short* sim   = (unsigned short*)alloc((size_t)NN * NN * 2);   // 128 MB
  unsigned short* abin  = (unsigned short*)alloc((size_t)NN * NN * 2);   // 128 MB, L2-resident
  unsigned int*   hist  = (unsigned int*)alloc(16384 * 4);
  unsigned int*   sel   = (unsigned int*)alloc(16);
  unsigned int*   rcnt  = (unsigned int*)alloc(16);
  unsigned int*   epsB  = (unsigned int*)alloc(16);
  float*          degI  = (float*)alloc((size_t)NN * 4);
  unsigned short* xb    = (unsigned short*)alloc((size_t)NN * DIN * 2);
  unsigned short* xnb   = (unsigned short*)alloc((size_t)NN * DIN * 2);
  unsigned short* w_inT = (unsigned short*)alloc((size_t)HH * DIN * 2);
  unsigned short* h1wlT = (unsigned short*)alloc((size_t)HH * HH * 2);
  unsigned short* h1wrT = (unsigned short*)alloc((size_t)HH * HH * 2);
  unsigned short* h2wlT = (unsigned short*)alloc((size_t)HH * HH * 2);
  unsigned short* h2wrT = (unsigned short*)alloc((size_t)HH * HH * 2);
  unsigned short* o1wlT = (unsigned short*)alloc((size_t)HH * 2);
  unsigned short* o1wrT = (unsigned short*)alloc((size_t)HH * 2);
  unsigned short* oscT  = (unsigned short*)alloc((size_t)HH * 2);
  unsigned short* act1  = (unsigned short*)alloc((size_t)NN * HH * 2);
  unsigned short* act2  = (unsigned short*)alloc((size_t)NN * HH * 2);
  unsigned short* act3  = (unsigned short*)alloc((size_t)NN * HH * 2);
  unsigned short* magg  = (unsigned short*)alloc((size_t)NN * HH * 2);
  unsigned short* hT    = (unsigned short*)alloc((size_t)HH * NN * 2);
  float*          accF  = (float*)alloc((size_t)NN * HH * 4);
  float*          oval  = (float*)alloc((size_t)NN * 4);
  float*          scv   = (float*)alloc((size_t)NN * 4);

  const unsigned long long nTot = (unsigned long long)NN * NN;
  const unsigned long long KRANK =
      (unsigned long long)(0.95 * (double)(nTot - 1) + 0.5);  // jnp.quantile "nearest"

  rownorm_k<<<NN, DIN, 0, stream>>>(x, xb, xnb);

  convt_w_k<<<(DIN * HH + 255) / 256, 256, 0, stream>>>(w_in, w_inT, DIN, HH);
  convt_w_k<<<(HH * HH + 255) / 256, 256, 0, stream>>>(h1_wl, h1wlT, HH, HH);
  convt_w_k<<<(HH * HH + 255) / 256, 256, 0, stream>>>(h1_wr, h1wrT, HH, HH);
  convt_w_k<<<(HH * HH + 255) / 256, 256, 0, stream>>>(h2_wl, h2wlT, HH, HH);
  convt_w_k<<<(HH * HH + 255) / 256, 256, 0, stream>>>(h2_wr, h2wrT, HH, HH);
  convt_w_k<<<(HH + 255) / 256, 256, 0, stream>>>(o1_wl, o1wlT, HH, 1);
  convt_w_k<<<(HH + 255) / 256, 256, 0, stream>>>(o1_wr, o1wrT, HH, 1);
  convt_w_k<<<(HH + 255) / 256, 256, 0, stream>>>(osc_w, oscT, HH, 1);

  // sim = |xn @ xn^T| (WMMA bf16, K=128)
  wmma_gemm_k<F_ABS><<<dim3(NN / 128, NN / 16), 256, 0, stream>>>(
      xnb, xnb, NN, NN, DIN, nullptr, nullptr, nullptr, sim, nullptr);

  // exact bf16 alpha-quantile via monotone-bit-pattern rank select
  zero_u32_k<<<(16384 + 255) / 256, 256, 0, stream>>>(hist, 16384);
  zero_u32_k<<<1, 32, 0, stream>>>(rcnt, 4);
  hist_k<<<1024, 256, 0, stream>>>(sim, nTot, hist);
  coarse_scan_k<<<1, 1, 0, stream>>>(hist, sel, KRANK);
  refine_count_k<<<1024, 256, 0, stream>>>(sim, nTot, sel, rcnt);
  final_eps_k<<<1, 1, 0, stream>>>(sel, rcnt, epsB);

  // materialize binary adjacency once; all aggregation GEMMs become pure dense WMMA
  binarize_k<<<4096, 256, 0, stream>>>(sim, abin, epsB, nTot);
  deginv_k<<<NN, 256, 0, stream>>>(sim, epsB, degI, NN);

  // input layer: act1 = relu(x @ w_in + b_in)
  wmma_gemm_k<F_BIAS | F_RELU><<<dim3(HH / 128, NN / 16), 256, 0, stream>>>(
      xb, w_inT, NN, HH, DIN, b_in, nullptr, nullptr, act1, nullptr);

  const dim3 gS(HH / 128, NN / 16);
  const dim3 gA(HH / 128, NN / AROWS);
  const dim3 gT(HH / 32, NN / 32), bT(32, 8);

  // SAGE layer 1
  transpose_k<<<gT, bT, 0, stream>>>(act1, hT, NN, HH);
  adj_gemm_lds_k<<<gA, 256, 0, stream>>>(abin, hT, HH, NN, degI, magg);
  wmma_gemm_k<F_BIAS | F_OUTF32><<<gS, 256, 0, stream>>>(
      magg, h1wlT, NN, HH, HH, h1_bl, nullptr, nullptr, nullptr, accF);
  wmma_gemm_k<F_ACC | F_RELU><<<gS, 256, 0, stream>>>(
      act1, h1wrT, NN, HH, HH, nullptr, accF, nullptr, act2, nullptr);

  // SAGE layer 2 + residual(act1)
  transpose_k<<<gT, bT, 0, stream>>>(act2, hT, NN, HH);
  adj_gemm_lds_k<<<gA, 256, 0, stream>>>(abin, hT, HH, NN, degI, magg);
  wmma_gemm_k<F_BIAS | F_OUTF32><<<gS, 256, 0, stream>>>(
      magg, h2wlT, NN, HH, HH, h2_bl, nullptr, nullptr, nullptr, accF);
  wmma_gemm_k<F_ACC | F_RES | F_RELU><<<gS, 256, 0, stream>>>(
      act2, h2wrT, NN, HH, HH, nullptr, accF, act1, act3, nullptr);

  // output head
  transpose_k<<<gT, bT, 0, stream>>>(act3, hT, NN, HH);
  adj_gemm_lds_k<<<gA, 256, 0, stream>>>(abin, hT, HH, NN, degI, magg);
  out1_k<<<NN, HH, 0, stream>>>(magg, act3, o1wlT, o1wrT, oscT, o1_bl, osc_b, oval, scv);
  final_k<<<NN, 256, 0, stream>>>(abin, oval, degI, o2_wl, o2_bl, o2_wr, scv, out, NN);
}
